// EmbeddingGCN_21878563406445
// MI455X (gfx1250) — matrix operand order repaired and checked by reading, not compiled, not run on hardware
//
#include <hip/hip_runtime.h>
#include <hip/hip_bf16.h>

typedef __attribute__((ext_vector_type(16))) _Float16 v16h;
typedef __attribute__((ext_vector_type(8)))  _Float16 v8h;
typedef __attribute__((ext_vector_type(8)))  float    v8f;

#define T_DIM 8
#define N_DIM 10000
#define F_DIM 64
#define E_NUM 1000000
#define NNZ_NUM 1280000
#define TN_DIM (T_DIM * N_DIM)   // 80000
#define NF_DIM (N_DIM * F_DIM)   // 640000

// ---------------------------------------------------------------------------
// Kernel 1: Xt = M @ X  along time.  One thread per (n,f) column of the
// (8 x 640000) GEMM; M is wave-uniform (scalar loads).
// ---------------------------------------------------------------------------
__global__ void k_time_fwd(const float* __restrict__ M,
                           const float* __restrict__ X,
                           float* __restrict__ Xt) {
    int j = blockIdx.x * blockDim.x + threadIdx.x;
    if (j >= NF_DIM) return;
    float x[T_DIM];
#pragma unroll
    for (int k = 0; k < T_DIM; ++k) x[k] = X[(long)k * NF_DIM + j];
#pragma unroll
    for (int t = 0; t < T_DIM; ++t) {
        float acc = 0.f;
#pragma unroll
        for (int k = 0; k < T_DIM; ++k) acc += M[t * T_DIM + k] * x[k];
        Xt[(long)t * NF_DIM + j] = acc;
    }
}

// ---------------------------------------------------------------------------
// Zero a float buffer (float4-wide).
// ---------------------------------------------------------------------------
__global__ void k_zero4(float4* __restrict__ p, int n4) {
    int i = blockIdx.x * blockDim.x + threadIdx.x;
    if (i < n4) p[i] = make_float4(0.f, 0.f, 0.f, 0.f);
}

// ---------------------------------------------------------------------------
// Kernel 2: COO SpMM scatter:  AtXt[at_dst] += at_val * Xt[at_src]
// One wave per nnz (32 lanes x float2 = 64 features); COO index/value
// streams are touched once -> non-temporal loads.  Atomics resolve in L2.
// ---------------------------------------------------------------------------
__global__ void k_spmm_scatter(const float* __restrict__ at_val,
                               const int* __restrict__ at_src,
                               const int* __restrict__ at_dst,
                               const float* __restrict__ Xt,
                               float* __restrict__ AtXt) {
    long tid = (long)blockIdx.x * blockDim.x + threadIdx.x;
    if (tid >= (long)NNZ_NUM * 32) return;
    int nz = (int)(tid >> 5);
    int f2 = (int)(tid & 31) * 2;
    float v  = __builtin_nontemporal_load(at_val + nz);
    int   s  = __builtin_nontemporal_load(at_src + nz);
    int   d  = __builtin_nontemporal_load(at_dst + nz);
    const float2 x = *(const float2*)(Xt + (long)s * F_DIM + f2);
    float* dp = AtXt + (long)d * F_DIM + f2;
    atomicAdd(dp,     v * x.x);
    atomicAdd(dp + 1, v * x.y);
}

// ---------------------------------------------------------------------------
// f32 -> f16 elementwise convert (row-major buffer for WMMA A-tiles).
// ---------------------------------------------------------------------------
__global__ void k_cvt_f16(const float* __restrict__ src,
                          _Float16* __restrict__ dst, int n) {
    int i = blockIdx.x * blockDim.x + threadIdx.x;
    if (i < n) dst[i] = (_Float16)src[i];
}

// ---------------------------------------------------------------------------
// Pack a row-major f32 K x 64 matrix into the per-lane WMMA B layout.
// Packed element r = ((kt*nt_n + nt)*32 + lane)*16 + e  corresponds to
// source element  [kt*32 + 8*(lane>>4) + (e&7) + 16*(e>>3)][nt*16 + (lane&15)].
// ---------------------------------------------------------------------------
__global__ void k_pack_b(const float* __restrict__ src,
                         _Float16* __restrict__ dst,
                         int ld, int nt_n,
                         long sstride, long dstride,
                         int per_b, int total) {
    int idx = blockIdx.x * blockDim.x + threadIdx.x;
    if (idx >= total) return;
    int b    = idx / per_b;
    int r    = idx - b * per_b;
    int e    = r & 15;
    int lane = (r >> 4) & 31;
    int tile = r >> 9;
    int nt   = tile % nt_n;
    int kt   = tile / nt_n;
    int K    = kt * 32 + ((lane >> 4) << 3) + (e & 7) + ((e >> 3) << 4);
    int nn   = nt * 16 + (lane & 15);
    dst[b * dstride + r] = (_Float16)src[b * sstride + (long)K * ld + nn];
}

// ---------------------------------------------------------------------------
// Kernel 3: AtXtWt[t] = AtXt[t] @ W[t]  via v_wmma_f32_16x16x32_f16.
// One wave per 16-row tile (K=64 -> 2 k-steps, N=64 -> 4 col tiles).
// ---------------------------------------------------------------------------
__global__ void __launch_bounds__(256)
k_gemm_w(const _Float16* __restrict__ A16,
         const _Float16* __restrict__ Wpk,
         float* __restrict__ Out) {
    int lane = threadIdx.x & 31;
    int tile = (int)((blockIdx.x * blockDim.x + threadIdx.x) >> 5);
    if (tile >= TN_DIM / 16) return;
    int r0 = tile * 16;
    int t  = r0 / N_DIM;           // 16 | N_DIM, tile never straddles t
    int m  = lane & 15;
    int kb = (lane >> 4) * 8;
    const _Float16* arow = A16 + (long)(r0 + m) * F_DIM;
    const _Float16* bp   = Wpk + (long)t * 4096;

    v8f c[4];
#pragma unroll
    for (int nt = 0; nt < 4; ++nt) c[nt] = (v8f){0.f,0.f,0.f,0.f,0.f,0.f,0.f,0.f};

#pragma unroll
    for (int kt = 0; kt < 2; ++kt) {
        v8h a1 = *(const v8h*)(arow + kt * 32 + kb);
        v8h a2 = *(const v8h*)(arow + kt * 32 + kb + 16);
        v16h a;
#pragma unroll
        for (int i = 0; i < 8; ++i) { a[i] = a1[i]; a[i + 8] = a2[i]; }
#pragma unroll
        for (int nt = 0; nt < 4; ++nt) {
            v16h b = *(const v16h*)(bp + ((kt * 4 + nt) * 32 + lane) * 16);
            c[nt] = __builtin_amdgcn_wmma_f32_16x16x32_f16(
                false, a, false, b, (short)0, c[nt], false, false);
        }
    }
    int ro = (lane >> 4) * 8;
#pragma unroll
    for (int nt = 0; nt < 4; ++nt)
#pragma unroll
        for (int r = 0; r < 8; ++r)
            Out[(long)(r0 + ro + r) * F_DIM + nt * 16 + (lane & 15)] = c[nt][r];
}

// ---------------------------------------------------------------------------
// Kernel 4: Y = Minv @ AtXtWt along time, converted to f16 for edge GEMM.
// ---------------------------------------------------------------------------
__global__ void k_time_inv_f16(const float* __restrict__ Minv,
                               const float* __restrict__ A,
                               _Float16* __restrict__ Yh) {
    int j = blockIdx.x * blockDim.x + threadIdx.x;
    if (j >= NF_DIM) return;
    float x[T_DIM];
#pragma unroll
    for (int k = 0; k < T_DIM; ++k) x[k] = A[(long)k * NF_DIM + j];
#pragma unroll
    for (int t = 0; t < T_DIM; ++t) {
        float acc = 0.f;
#pragma unroll
        for (int k = 0; k < T_DIM; ++k) acc += Minv[t * T_DIM + k] * x[k];
        Yh[(long)t * NF_DIM + j] = (_Float16)acc;
    }
}

// ---------------------------------------------------------------------------
// Kernel 5: out = concat(Y[src], Y[trg]) @ U  -- the dominant GEMM.
// One wave per 16 edges: A is the virtual 16x128 concat row (each 16-byte
// A-chunk falls entirely inside one of the two Y rows, so it's a per-lane
// pointer select), B is pre-packed U, K=128 -> 4 k-steps, N=64 -> 4 tiles,
// 16 WMMAs per wave.  The 256 MB output is streamed once -> non-temporal
// stores keep Yh/Upk resident in the 192 MB L2 for the gathers.
// ---------------------------------------------------------------------------
__global__ void __launch_bounds__(256)
k_edge_gemm(const int* __restrict__ edges,
            const _Float16* __restrict__ Yh,
            const _Float16* __restrict__ Upk,
            float* __restrict__ out) {
    int lane = threadIdx.x & 31;
    int tile = (int)((blockIdx.x * blockDim.x + threadIdx.x) >> 5);
    if (tile >= E_NUM / 16) return;
    int e0 = tile << 4;
    int m  = lane & 15;
    int e  = e0 + m;
    int t  = __builtin_nontemporal_load(edges + e);
    int sn = __builtin_nontemporal_load(edges + E_NUM + e);
    int tn = __builtin_nontemporal_load(edges + 2 * E_NUM + e);
    const _Float16* srow = Yh + (long)(t * N_DIM + sn) * F_DIM;
    const _Float16* trow = Yh + (long)(t * N_DIM + tn) * F_DIM;
    int kb = (lane >> 4) * 8;

    v8f c[4];
#pragma unroll
    for (int nt = 0; nt < 4; ++nt) c[nt] = (v8f){0.f,0.f,0.f,0.f,0.f,0.f,0.f,0.f};

#pragma unroll
    for (int kt = 0; kt < 4; ++kt) {
        int k1 = kt * 32 + kb;
        int k2 = k1 + 16;
        const _Float16* p1 = (k1 < F_DIM) ? (srow + k1) : (trow + (k1 - F_DIM));
        const _Float16* p2 = (k2 < F_DIM) ? (srow + k2) : (trow + (k2 - F_DIM));
        v8h a1 = *(const v8h*)p1;
        v8h a2 = *(const v8h*)p2;
        v16h a;
#pragma unroll
        for (int i = 0; i < 8; ++i) { a[i] = a1[i]; a[i + 8] = a2[i]; }
#pragma unroll
        for (int nt = 0; nt < 4; ++nt) {
            v16h b = *(const v16h*)(Upk + ((kt * 4 + nt) * 32 + lane) * 16);
            c[nt] = __builtin_amdgcn_wmma_f32_16x16x32_f16(
                false, a, false, b, (short)0, c[nt], false, false);
        }
    }
    int ro = (lane >> 4) * 8;
#pragma unroll
    for (int nt = 0; nt < 4; ++nt)
#pragma unroll
        for (int r = 0; r < 8; ++r)
            __builtin_nontemporal_store(
                c[nt][r],
                out + (long)(e0 + ro + r) * F_DIM + nt * 16 + (lane & 15));
}

// ---------------------------------------------------------------------------
extern "C" void kernel_launch(void* const* d_in, const int* in_sizes, int n_in,
                              void* d_out, int out_size, void* d_ws, size_t ws_size,
                              hipStream_t stream) {
    const float* X      = (const float*)d_in[0];
    const float* M      = (const float*)d_in[1];
    const float* Minv   = (const float*)d_in[2];
    const float* W      = (const float*)d_in[3];
    const float* U      = (const float*)d_in[4];
    const float* at_val = (const float*)d_in[5];
    const int*   at_src = (const int*)d_in[6];
    const int*   at_dst = (const int*)d_in[7];
    const int*   edges  = (const int*)d_in[8];
    float*       out    = (float*)d_out;

    char* ws = (char*)d_ws;
    // Workspace layout (51.3 MB, regions reused once their producer is done):
    float*    Xt     = (float*)(ws + 0);           // 5,120,000 f32
    float*    AtXt   = (float*)(ws + 20480000);    // 5,120,000 f32
    _Float16* AtXt16 = (_Float16*)(ws + 40960000); // 5,120,000 f16
    _Float16* Wpk    = (_Float16*)(ws + 51200000); // 32768 f16 (8 x 2x4 B-tiles)
    _Float16* Upk    = (_Float16*)(ws + 51265536); // 8192  f16 (4x4 B-tiles)
    float*    AtXtWt = Xt;                         // reuse: Xt dead after scatter
    _Float16* Yh     = (_Float16*)AtXt;            // reuse: AtXt dead after cvt

    const int B = 256;

    // 1) Xt = M @ X
    k_time_fwd<<<(NF_DIM + B - 1) / B, B, 0, stream>>>(M, X, Xt);

    // pack B-operands (independent of everything else)
    k_pack_b<<<(8192 + B - 1) / B, B, 0, stream>>>(U, Upk, F_DIM, 4, 0, 0, 8192, 8192);
    k_pack_b<<<(32768 + B - 1) / B, B, 0, stream>>>(W, Wpk, F_DIM, 4,
                                                    (long)F_DIM * F_DIM, 4096, 4096, 32768);

    // 2) AtXt = segment_sum(at_val * Xt[at_src])
    int n4 = (TN_DIM * F_DIM) / 4;
    k_zero4<<<(n4 + B - 1) / B, B, 0, stream>>>((float4*)AtXt, n4);
    long nsc = (long)NNZ_NUM * 32;
    k_spmm_scatter<<<(int)((nsc + B - 1) / B), B, 0, stream>>>(at_val, at_src, at_dst, Xt, AtXt);

    // 3) AtXtWt[t] = AtXt[t] @ W[t]   (WMMA)
    int nelem = TN_DIM * F_DIM;
    k_cvt_f16<<<(nelem + B - 1) / B, B, 0, stream>>>(AtXt, AtXt16, nelem);
    int wtiles = TN_DIM / 16;                       // 5000 waves
    k_gemm_w<<<(wtiles * 32 + B - 1) / B, B, 0, stream>>>(AtXt16, Wpk, AtXtWt);

    // 4) Y = Minv @ AtXtWt  -> f16
    k_time_inv_f16<<<(NF_DIM + B - 1) / B, B, 0, stream>>>(Minv, AtXtWt, Yh);

    // 5) out = [Y[src] | Y[trg]] @ U   (WMMA, dominant kernel)
    int etiles = E_NUM / 16;                        // 62500 waves
    k_edge_gemm<<<(etiles * 32 + B - 1) / B, B, 0, stream>>>(edges, Yh, Upk, out);
}
